// TokenClassifier_17248588661482
// MI455X (gfx1250) — matrix operand order repaired and checked by reading, not compiled, not run on hardware
//
#include <hip/hip_runtime.h>
#include <hip/hip_bf16.h>
#include <math.h>

typedef __attribute__((ext_vector_type(16))) __bf16 v16bf;
typedef __attribute__((ext_vector_type(8)))  __bf16 v8bf;
typedef __attribute__((ext_vector_type(8)))  float  v8f;

#define Bq 2
#define Sq 4096
#define Dq 768
#define Hh 12
#define DHq 64
#define Wn 128
#define Gg 16
#define Lc 24
#define NC 32            // Sq / Wn
#define KC 416           // 16 global + 384 band + 16 pad (multiple of 32)
#define NEGV (-30000.0f)

// ---- CDNA5 async global->LDS (ASYNCcnt) via inline asm (bypasses builtin signature) ----
#define USE_ASYNC_LDS 1

#if USE_ASYNC_LDS
// GLOBAL_LOAD_ASYNC_TO_LDS_B128: VDST = LDS byte address (low 32 bits of flat
// shared pointer == LDS offset per aperture rules), VADDR = 64-bit global addr.
static __device__ inline void async_cp16(__bf16* lds_dst, const __bf16* gsrc) {
  unsigned lds = (unsigned)(size_t)lds_dst;
  unsigned long long ga = (unsigned long long)(size_t)gsrc;
  asm volatile("global_load_async_to_lds_b128 %0, %1, off"
               :: "v"(lds), "v"(ga) : "memory");
}
static __device__ inline void wait_async_le1() {
  asm volatile("s_wait_asynccnt 1" ::: "memory");
}
static __device__ inline void wait_async_le0() {
  asm volatile("s_wait_asynccnt 0" ::: "memory");
}
#endif

static __device__ inline v8f zero8() {
  v8f z;
#pragma unroll
  for (int i = 0; i < 8; ++i) z[i] = 0.f;
  return z;
}

static __device__ inline v16bf zero16() {
  v16bf z;
#pragma unroll
  for (int i = 0; i < 16; ++i) z[i] = (__bf16)0.f;
  return z;
}

static __device__ inline v16bf cat8(v8bf lo, v8bf hi) {
  return __builtin_shufflevector(lo, hi, 0, 1, 2, 3, 4, 5, 6, 7,
                                 8, 9, 10, 11, 12, 13, 14, 15);
}

static __device__ inline v8bf ldv8(const __bf16* p) { return *(const v8bf*)p; }

static __device__ inline v8f wmma_bf16(v16bf a, v16bf b, v8f c) {
  return __builtin_amdgcn_wmma_f32_16x16x32_bf16(false, a, false, b, (short)0, c,
                                                 false, false);
}

// A fragment (16x32 bf16, row-major M x K, stride ld). Two 16B loads per lane.
// lane&15 = row M; lane>>4 selects K halves {0..15} vs {16..31}.
static __device__ inline v16bf load_a16(const __bf16* p, int ld, int lane) {
  const __bf16* r = p + (lane & 15) * ld + ((lane >> 4) << 4);
  return cat8(ldv8(r), ldv8(r + 8));
}

// B fragment from a TRANSPOSED operand (rows = N, cols = K, stride ld).
// lane&15 = column N; lanes 0-15 hold K=0..15, lanes 16-31 hold K=16..31.
// Per lane: 16 contiguous bf16 = two 16B loads.
static __device__ inline v16bf load_bt16(const __bf16* p, int ld, int lane) {
  const __bf16* r = p + (lane & 15) * ld + ((lane >> 4) << 4);
  return cat8(ldv8(r), ldv8(r + 8));
}

// ---------------- elementwise helpers ----------------
// cast + transpose: in R x C (f32, row-major) -> out C x R (bf16)
__global__ void k_cast_transpose(const float* __restrict__ in, __bf16* __restrict__ out,
                                 int R, int C) {
  int total = R * C;
  for (int i = blockIdx.x * blockDim.x + threadIdx.x; i < total; i += gridDim.x * blockDim.x) {
    int r = i / C, c = i - r * C;
    out[(long)c * R + r] = (__bf16)in[i];
  }
}

__global__ void k_embed(const int* __restrict__ ids, const float* __restrict__ emb,
                        __bf16* __restrict__ x, int rows) {
  int total = rows * Dq;
  for (int i = blockIdx.x * blockDim.x + threadIdx.x; i < total; i += gridDim.x * blockDim.x) {
    int r = i / Dq, d = i - r * Dq;
    x[i] = (__bf16)emb[(long)ids[r] * Dq + d];
  }
}

// (B*S, D) -> head-major transposed (B*H*DH, S):  out[((b*H+h)*64+d)*S + s]
__global__ void k_transpose_heads(const __bf16* __restrict__ in, __bf16* __restrict__ out) {
  int total = Bq * Sq * Dq;
  for (int i = blockIdx.x * blockDim.x + threadIdx.x; i < total; i += gridDim.x * blockDim.x) {
    int row = i / Dq, dcol = i - row * Dq;
    int b = row / Sq, s = row - b * Sq;
    int h = dcol >> 6, d = dcol & 63;
    out[((long)(b * Hh + h) * DHq + d) * Sq + s] = in[i];
  }
}

// ---------------- generic bf16 GEMM: C = scale * (A @ B),  B given TRANSPOSED ----------------
// A: M x K, Bt: N x K, C: M x N (all bf16, f32 accumulation).
// Block: 256 threads = 8 waves; block tile 128(M) x 64(N); wave tile 16 x 64.
// B tile staged in LDS; double-buffered async global->LDS copy.
__global__ void __launch_bounds__(256) k_gemm_bf16(
    const __bf16* __restrict__ A, const __bf16* __restrict__ Bt,
    __bf16* __restrict__ C, int M, int N, int K, float scale) {
  __shared__ __bf16 Bts[2][64 * 32];  // [buf][n_local][k], rows 64B -> b128 friendly
  const int lane = threadIdx.x & 31, wave = threadIdx.x >> 5;
  const int m0 = blockIdx.x * 128 + wave * 16;
  const int n0 = blockIdx.y * 64;
  const int row = threadIdx.x >> 2, ch = threadIdx.x & 3;  // 64 rows x 4 x 16B chunks
  const __bf16* src = Bt + (long)(n0 + row) * K + ch * 8;
  __bf16* dst0 = &Bts[0][row * 32 + ch * 8];
  __bf16* dst1 = &Bts[1][row * 32 + ch * 8];
  v8f acc[4];
#pragma unroll
  for (int t = 0; t < 4; ++t) acc[t] = zero8();

#if USE_ASYNC_LDS
  async_cp16(dst0, src);                 // prologue: tile 0 in flight
  for (int k0 = 0; k0 < K; k0 += 32) {
    const bool more = (k0 + 32 < K);
    __bf16* cur = ((k0 >> 5) & 1) ? &Bts[1][0] : &Bts[0][0];
    if (more) {
      async_cp16(((k0 >> 5) & 1) ? dst0 : dst1, src + k0 + 32);
      wait_async_le1();                  // current tile has landed
    } else {
      wait_async_le0();
    }
    __syncthreads();
    if (more)
      __builtin_prefetch(&A[(long)(m0 + (lane & 15)) * K + k0 + 32], 0, 1);
    v16bf a = load_a16(A + (long)m0 * K + k0, K, lane);
    v16bf bf[4];
#pragma unroll
    for (int t = 0; t < 4; ++t) bf[t] = load_bt16(cur + t * 16 * 32, 32, lane);
#pragma unroll
    for (int t = 0; t < 4; ++t) acc[t] = wmma_bf16(a, bf[t], acc[t]);
    __syncthreads();   // all reads done before this buffer is overwritten again
  }
#else
  for (int k0 = 0; k0 < K; k0 += 32) {
    *(v8bf*)dst0 = ldv8(src + k0);
    __syncthreads();
    if (k0 + 64 < K)
      __builtin_prefetch(&A[(long)(m0 + (lane & 15)) * K + k0 + 64], 0, 1);
    v16bf a = load_a16(A + (long)m0 * K + k0, K, lane);
    v16bf bf[4];
#pragma unroll
    for (int t = 0; t < 4; ++t) bf[t] = load_bt16(&Bts[0][t * 16 * 32], 32, lane);
#pragma unroll
    for (int t = 0; t < 4; ++t) acc[t] = wmma_bf16(a, bf[t], acc[t]);
    __syncthreads();
  }
#endif
  const int col = lane & 15, mb = (lane >> 4) << 3;
#pragma unroll
  for (int t = 0; t < 4; ++t)
#pragma unroll
    for (int r = 0; r < 8; ++r)
      C[(long)(m0 + mb + r) * N + n0 + t * 16 + col] = (__bf16)(acc[t][r] * scale);
}

// ---------------- banded local attention, one block per (b,h,chunk) ----------------
// Combined key space: cols 0..15 = global-prefix keys, 16..399 = 3W band, 400..415 pad.
__global__ void __launch_bounds__(256, 1) k_band_attn(
    const __bf16* __restrict__ qb, const __bf16* __restrict__ kb,
    const __bf16* __restrict__ vT, const int* __restrict__ am,
    __bf16* __restrict__ ctx) {
  extern __shared__ char smem[];
  float*  sS = (float*)smem;                    // [128][KC] scores f32
  __bf16* sP = (__bf16*)(smem + 128 * KC * 4);  // [128][KC] probs bf16
  const int lane = threadIdx.x & 31, wave = threadIdx.x >> 5;
  const int blk = blockIdx.x;
  const int c = blk % NC, h = (blk / NC) % Hh, b = blk / (NC * Hh);
  const long base = (long)b * Sq;
  const int hoff = h * DHq;
  const long vrow0 = (long)(b * Hh + h) * DHq;  // row base in vT

  // phase 1: scores = q_chunk(128x64) @ Kc^T(64x416)
  {
    const __bf16* qrow = qb + (base + (long)c * Wn + wave * 16) * Dq + hoff;
    v16bf a0 = load_a16(qrow, Dq, lane);
    v16bf a1 = load_a16(qrow + 32, Dq, lane);
    const int col = lane & 15, mb = (lane >> 4) << 3;
    for (int nt = 0; nt < KC / 16; ++nt) {
      int n = nt * 16 + col;
      int krow;
      if (n < Gg) krow = n;
      else if (n < Gg + 3 * Wn) {
        int ka = c * Wn + (n - Gg) - Wn;
        krow = (ka >= 0 && ka < Sq) ? ka : -1;
      } else krow = -1;
      v16bf b0, b1;
      if (krow >= 0) {
        // per-lane contiguous along DH: 2x(2x16B) vector loads
        const __bf16* kr = kb + (base + krow) * Dq + hoff + ((lane >> 4) << 4);
        b0 = cat8(ldv8(kr), ldv8(kr + 8));
        b1 = cat8(ldv8(kr + 32), ldv8(kr + 40));
      } else {
        b0 = zero16();
        b1 = zero16();
      }
      v8f accd = zero8();
      accd = wmma_bf16(a0, b0, accd);
      accd = wmma_bf16(a1, b1, accd);
#pragma unroll
      for (int r = 0; r < 8; ++r)
        sS[(wave * 16 + mb + r) * KC + nt * 16 + col] = accd[r];
    }
  }
  __syncthreads();

  // phase 2: validity mask + attention-mask additive bias
  for (int idx = threadIdx.x; idx < 128 * KC; idx += 256) {
    int m = idx / KC, n = idx - m * KC;
    int q_abs = c * Wn + m;
    float sc = sS[idx];
    if (n < Gg) {
      sc += (1.0f - (float)am[base + n]) * NEGV;
    } else if (n < Gg + 3 * Wn) {
      int ka = c * Wn + (n - Gg) - Wn;
      int rel = ka - q_abs;
      bool valid = (rel <= Wn) && (rel >= -Wn) && (ka >= Gg) && (ka < Sq);
      float madd = (ka >= 0 && ka < Sq) ? (1.0f - (float)am[base + ka]) * NEGV : 0.f;
      sc = (valid ? sc : NEGV) + madd;
    } else {
      sc = NEGV;  // pad columns
    }
    sS[idx] = sc;
  }
  __syncthreads();

  // phase 3: row softmax -> bf16 probs
  if (threadIdx.x < 128) {
    const float* row = sS + threadIdx.x * KC;
    __bf16* prow = sP + threadIdx.x * KC;
    float mx = -3.4e38f;
    for (int n = 0; n < KC; ++n) mx = fmaxf(mx, row[n]);
    float sum = 0.f;
    for (int n = 0; n < KC; ++n) sum += expf(row[n] - mx);
    float inv = 1.0f / sum;
    for (int n = 0; n < KC; ++n) prow[n] = (__bf16)(expf(row[n] - mx) * inv);
  }
  __syncthreads();

  // phase 4: out = P(128x416) @ Vc(416x64), V read from head-major transposed vT
  {
    const int col = lane & 15, mb = (lane >> 4) << 3;
    v8f acc[4];
#pragma unroll
    for (int t = 0; t < 4; ++t) acc[t] = zero8();
    for (int k0 = 0; k0 < KC; k0 += 32) {
      v16bf a = load_a16(sP + (wave * 16) * KC + k0, KC, lane);
      int kk0 = k0 + ((lane >> 4) << 4);  // 16 consecutive combined-key rows
#pragma unroll
      for (int t = 0; t < 4; ++t) {
        const __bf16* vrow = vT + (vrow0 + t * 16 + col) * Sq;
        v16bf bvec;
        if (kk0 < Gg) {  // kk0 == 0: keys 0..15 (global prefix)
          bvec = cat8(ldv8(vrow), ldv8(vrow + 8));
        } else if (kk0 < Gg + 3 * Wn) {
          int ka0 = c * Wn + kk0 - Gg - Wn;
          if (ka0 >= 0 && ka0 + 16 <= Sq) {
            bvec = cat8(ldv8(vrow + ka0), ldv8(vrow + ka0 + 8));
          } else {
#pragma unroll
            for (int i = 0; i < 16; ++i) {
              int ka = ka0 + i;
              bvec[i] = (ka >= 0 && ka < Sq) ? vrow[ka] : (__bf16)0.f;
            }
          }
        } else {
          bvec = zero16();
        }
        acc[t] = wmma_bf16(a, bvec, acc[t]);
      }
    }
#pragma unroll
    for (int t = 0; t < 4; ++t)
#pragma unroll
      for (int r = 0; r < 8; ++r)
        ctx[(base + (long)c * Wn + wave * 16 + mb + r) * Dq + hoff + t * 16 + col] =
            (__bf16)acc[t][r];
  }
}

// ---------------- global attention for first G tokens, one block per (b,h) ----------------
__global__ void __launch_bounds__(256, 1) k_global_attn(
    const __bf16* __restrict__ qg, const __bf16* __restrict__ kg,
    const __bf16* __restrict__ vgT, const int* __restrict__ am,
    __bf16* __restrict__ ctx) {
  extern __shared__ char smem[];
  __bf16* sS = (__bf16*)smem;  // [16][Sq+8] bf16 scores -> probs (in place)
  const int LD = Sq + 8;
  const int lane = threadIdx.x & 31, wave = threadIdx.x >> 5;
  const int h = blockIdx.x % Hh, b = blockIdx.x / Hh;
  const long base = (long)b * Sq;
  const int hoff = h * DHq;
  const long vrow0 = (long)(b * Hh + h) * DHq;

  // phase 1: scores = qg[:16] (16x64) @ kg^T (64xS), + mask bias
  {
    const __bf16* qrow = qg + base * Dq + hoff;
    v16bf a0 = load_a16(qrow, Dq, lane);
    v16bf a1 = load_a16(qrow + 32, Dq, lane);
    const int col = lane & 15, mb = (lane >> 4) << 3;
    for (int nt = wave; nt < Sq / 16; nt += 8) {
      int n = nt * 16 + col;
      const __bf16* kr = kg + (base + n) * Dq + hoff + ((lane >> 4) << 4);
      v16bf b0 = cat8(ldv8(kr), ldv8(kr + 8));
      v16bf b1 = cat8(ldv8(kr + 32), ldv8(kr + 40));
      v8f accd = zero8();
      accd = wmma_bf16(a0, b0, accd);
      accd = wmma_bf16(a1, b1, accd);
      float madd = (1.0f - (float)am[base + n]) * NEGV;
#pragma unroll
      for (int r = 0; r < 8; ++r)
        sS[(mb + r) * LD + n] = (__bf16)(accd[r] + madd);
    }
  }
  __syncthreads();

  // phase 2: row softmax (16 rows), in place
  if (threadIdx.x < 16) {
    __bf16* row = sS + threadIdx.x * LD;
    float mx = -3.4e38f;
    for (int n = 0; n < Sq; ++n) mx = fmaxf(mx, (float)row[n]);
    float sum = 0.f;
    for (int n = 0; n < Sq; ++n) sum += expf((float)row[n] - mx);
    float inv = 1.0f / sum;
    for (int n = 0; n < Sq; ++n) row[n] = (__bf16)(expf((float)row[n] - mx) * inv);
  }
  __syncthreads();

  // phase 3: out = P(16xS) @ vg(Sx64); waves 0..3 each own a 16-col tile
  if (wave < 4) {
    const int col = lane & 15, mb = (lane >> 4) << 3;
    const __bf16* vrow = vgT + (vrow0 + wave * 16 + col) * Sq;
    v8f acc = zero8();
    for (int k0 = 0; k0 < Sq; k0 += 32) {
      v16bf a = load_a16(sS + k0, LD, lane);
      int kk0 = k0 + ((lane >> 4) << 4);
      v16bf bvec = cat8(ldv8(vrow + kk0), ldv8(vrow + kk0 + 8));
      acc = wmma_bf16(a, bvec, acc);
    }
#pragma unroll
    for (int r = 0; r < 8; ++r)
      ctx[(base + mb + r) * Dq + hoff + wave * 16 + col] = (__bf16)acc[r];
  }
}

// ---------------- classifier: logits = proj @ Wc + bc, softmax over 24 ----------------
// wcT: Lc x Dq (transposed) so each lane reads a contiguous row.
__global__ void __launch_bounds__(256) k_classifier(
    const __bf16* __restrict__ xin, const __bf16* __restrict__ wcT,
    const float* __restrict__ bc, float* __restrict__ out, int rows) {
  const int lane = threadIdx.x & 31;
  const int row = blockIdx.x * 8 + (threadIdx.x >> 5);
  if (row >= rows) return;
  float logit = -3.4e38f;
  if (lane < Lc) {
    float s = bc[lane];
    const __bf16* xr = xin + (long)row * Dq;
    const __bf16* wr = wcT + (long)lane * Dq;
    for (int d = 0; d < Dq; d += 8) {
      v8bf xv = ldv8(xr + d);
      v8bf wv = ldv8(wr + d);
#pragma unroll
      for (int j = 0; j < 8; ++j) s += (float)xv[j] * (float)wv[j];
    }
    logit = s;
  }
  float mx = logit;
  for (int o = 16; o > 0; o >>= 1) mx = fmaxf(mx, __shfl_xor(mx, o, 32));
  float e = (lane < Lc) ? expf(logit - mx) : 0.f;
  float sum = e;
  for (int o = 16; o > 0; o >>= 1) sum += __shfl_xor(sum, o, 32);
  if (lane < Lc) out[(long)row * Lc + lane] = e / sum;
}

extern "C" void kernel_launch(void* const* d_in, const int* in_sizes, int n_in,
                              void* d_out, int out_size, void* d_ws, size_t ws_size,
                              hipStream_t stream) {
  (void)in_sizes; (void)n_in; (void)out_size; (void)ws_size;
  const int*   ids = (const int*)d_in[0];
  const int*   am  = (const int*)d_in[1];
  const float* emb = (const float*)d_in[2];
  const float* wsrc[7] = { (const float*)d_in[3],  // Wq
                           (const float*)d_in[4],  // Wk
                           (const float*)d_in[5],  // Wv
                           (const float*)d_in[7],  // Wqg
                           (const float*)d_in[8],  // Wkg
                           (const float*)d_in[9],  // Wvg
                           (const float*)d_in[6] };// Wo
  const float* Wc = (const float*)d_in[10];
  const float* bc = (const float*)d_in[11];
  float* out = (float*)d_out;

  const int M = Bq * Sq;  // 8192 token rows
  char* ws = (char*)d_ws;
  size_t off = 0;
  auto take = [&](size_t bytes) {
    char* p = ws + off;
    off = (off + bytes + 255) & ~(size_t)255;
    return p;
  };
  __bf16* xbf = (__bf16*)take((size_t)M * Dq * 2);
  __bf16* wtbf[7];
  for (int i = 0; i < 7; ++i) wtbf[i] = (__bf16*)take((size_t)Dq * Dq * 2);
  __bf16* wcT = (__bf16*)take((size_t)Dq * Lc * 2);
  __bf16* qb   = (__bf16*)take((size_t)M * Dq * 2);
  __bf16* kb   = (__bf16*)take((size_t)M * Dq * 2);
  __bf16* vb   = (__bf16*)take((size_t)M * Dq * 2);
  __bf16* qgb  = (__bf16*)take((size_t)M * Dq * 2);
  __bf16* kgb  = (__bf16*)take((size_t)M * Dq * 2);
  __bf16* vgb  = (__bf16*)take((size_t)M * Dq * 2);
  __bf16* vT   = (__bf16*)take((size_t)M * Dq * 2);   // (B*H*DH, S)
  __bf16* vgT  = (__bf16*)take((size_t)M * Dq * 2);   // (B*H*DH, S)
  __bf16* ctx  = (__bf16*)take((size_t)M * Dq * 2);
  __bf16* proj = (__bf16*)take((size_t)M * Dq * 2);

  for (int i = 0; i < 7; ++i)
    k_cast_transpose<<<dim3(512), dim3(256), 0, stream>>>(wsrc[i], wtbf[i], Dq, Dq);
  k_cast_transpose<<<dim3(72), dim3(256), 0, stream>>>(Wc, wcT, Dq, Lc);
  k_embed<<<dim3(2048), dim3(256), 0, stream>>>(ids, emb, xbf, M);

  const float scale = 0.125f;  // 1/sqrt(DH)
  dim3 gg(M / 128, Dq / 64);
  k_gemm_bf16<<<gg, 256, 0, stream>>>(xbf, wtbf[0], qb,  M, Dq, Dq, scale);
  k_gemm_bf16<<<gg, 256, 0, stream>>>(xbf, wtbf[1], kb,  M, Dq, Dq, 1.0f);
  k_gemm_bf16<<<gg, 256, 0, stream>>>(xbf, wtbf[2], vb,  M, Dq, Dq, 1.0f);
  k_gemm_bf16<<<gg, 256, 0, stream>>>(xbf, wtbf[3], qgb, M, Dq, Dq, scale);
  k_gemm_bf16<<<gg, 256, 0, stream>>>(xbf, wtbf[4], kgb, M, Dq, Dq, 1.0f);
  k_gemm_bf16<<<gg, 256, 0, stream>>>(xbf, wtbf[5], vgb, M, Dq, Dq, 1.0f);

  k_transpose_heads<<<dim3(2048), dim3(256), 0, stream>>>(vb, vT);
  k_transpose_heads<<<dim3(2048), dim3(256), 0, stream>>>(vgb, vgT);

  const size_t band_lds = (size_t)128 * KC * 4 + (size_t)128 * KC * 2;  // 312 KB < 320 KB WGP LDS
  k_band_attn<<<dim3(Bq * Hh * NC), 256, band_lds, stream>>>(qb, kb, vT, am, ctx);
  const size_t glob_lds = (size_t)16 * (Sq + 8) * 2;                    // ~128 KB
  k_global_attn<<<dim3(Bq * Hh), 256, glob_lds, stream>>>(qgb, kgb, vgT, am, ctx);

  k_gemm_bf16<<<gg, 256, 0, stream>>>(ctx, wtbf[6], proj, M, Dq, Dq, 1.0f);
  k_classifier<<<dim3(M / 8), 256, 0, stream>>>(proj, wcT, bc, out, M);
}
